// UberCRSN_23536420782604
// MI455X (gfx1250) — compile-verified, matrix operand order checked by reading, not run on hardware
//
#include <hip/hip_runtime.h>
#include <hip/hip_bf16.h>
#include <stdint.h>

// ---------------------------------------------------------------------------
// UberCRSN forward for MI455X (gfx1250): all GEMMs on bf16 WMMA with bf16
// operands prepared out-of-loop; async LDS staging for the big V=32000 GEMMs.
// ---------------------------------------------------------------------------

#define B_      128
#define D_      512
#define TWOD    1024
#define V_      32000
#define APAL    64
#define S0_     128
#define MAXREC_ 12
#define SMAX_   1664
#define TEMP_   0.8f
#define SCALE_  0.04419417382415922f  // 512^-0.5

typedef __bf16 bf16_t;
typedef __attribute__((ext_vector_type(16))) __bf16 v16bf;
typedef __attribute__((ext_vector_type(8)))  float  v8f;

// ------------------------------ helpers ------------------------------------

__device__ inline v8f wmma_bf16(v16bf a, v16bf b, v8f c) {
  return __builtin_amdgcn_wmma_f32_16x16x32_bf16(
      /*neg_a=*/false, a, /*neg_b=*/false, b,
      /*c_mod=*/(short)0, c, /*reuse_a=*/false, /*reuse_b=*/false);
}

// A/B fragment, row-major source with K contiguous. ISA 16-bit 16x32 layout:
// lanes 0-15 -> rows 0..15, K in {k0..k0+7} u {k0+16..k0+23};
// lanes 16-31 -> same rows, K shifted by +8.
__device__ inline v16bf frag_rm_f32(const float* base, int ld, int row0, int k0) {
  int lane = threadIdx.x & 31;
  const float* p = base + (size_t)(row0 + (lane & 15)) * ld + k0 + ((lane >> 4) << 3);
  v16bf f;
#pragma unroll
  for (int i = 0; i < 8; ++i) f[i] = (__bf16)p[i];
#pragma unroll
  for (int i = 0; i < 8; ++i) f[8 + i] = (__bf16)p[16 + i];
  return f;
}

__device__ inline v16bf frag_rm_bf16(const bf16_t* base, int ld, int row0, int k0) {
  int lane = threadIdx.x & 31;
  const bf16_t* p = base + (size_t)(row0 + (lane & 15)) * ld + k0 + ((lane >> 4) << 3);
  v16bf f;
#pragma unroll
  for (int i = 0; i < 8; ++i) f[i] = p[i];
#pragma unroll
  for (int i = 0; i < 8; ++i) f[8 + i] = p[16 + i];
  return f;
}

// B fragment from column-major views: element(k, col) = base[k*ld + col].
__device__ inline v16bf frag_cm_f32(const float* base, int ld, int col0, int k0) {
  int lane = threadIdx.x & 31;
  const float* p = base + (size_t)(k0 + ((lane >> 4) << 3)) * ld + col0 + (lane & 15);
  v16bf f;
#pragma unroll
  for (int i = 0; i < 8; ++i) f[i] = (__bf16)p[(size_t)i * ld];
#pragma unroll
  for (int i = 0; i < 8; ++i) f[8 + i] = (__bf16)p[(size_t)(16 + i) * ld];
  return f;
}

__device__ inline v16bf frag_cm_bf16(const bf16_t* base, int ld, int col0, int k0) {
  int lane = threadIdx.x & 31;
  const bf16_t* p = base + (size_t)(k0 + ((lane >> 4) << 3)) * ld + col0 + (lane & 15);
  v16bf f;
#pragma unroll
  for (int i = 0; i < 8; ++i) f[i] = p[(size_t)i * ld];
#pragma unroll
  for (int i = 0; i < 8; ++i) f[8 + i] = p[(size_t)(16 + i) * ld];
  return f;
}

// C tile store per ISA 32-bit C layout: lane n = col, vgpr r -> row r (+8 for hi lanes).
__device__ inline void store_tile(float* C, int ldc, int row0, int col0, v8f c,
                                  float scale, const float* bias) {
  int lane = threadIdx.x & 31;
  int r0 = row0 + ((lane >> 4) << 3);
  int cc = col0 + (lane & 15);
  float bv = bias ? bias[cc] : 0.f;
#pragma unroll
  for (int r = 0; r < 8; ++r)
    C[(size_t)(r0 + r) * ldc + cc] = c[r] * scale + bv;
}

__device__ inline void store_tile_bf16(bf16_t* C, int ldc, int row0, int col0, v8f c) {
  int lane = threadIdx.x & 31;
  int r0 = row0 + ((lane >> 4) << 3);
  int cc = col0 + (lane & 15);
#pragma unroll
  for (int r = 0; r < 8; ++r)
    C[(size_t)(r0 + r) * ldc + cc] = (bf16_t)c[r];
}

__device__ inline void async_b128(const void* gsrc, uint32_t lds_byte_addr) {
#if defined(__HIP_DEVICE_COMPILE__)
  asm volatile("global_load_async_to_lds_b128 %0, %1, off"
               :: "v"(lds_byte_addr), "v"(gsrc) : "memory");
#endif
}
__device__ inline void wait_async0() {
#if defined(__HIP_DEVICE_COMPILE__)
  asm volatile("s_wait_asynccnt 0" ::: "memory");
#endif
}

__device__ inline float block_reduce_sum(float v, float* sbuf) {
  int t = threadIdx.x;
  sbuf[t] = v; __syncthreads();
  for (int s = blockDim.x >> 1; s > 0; s >>= 1) {
    if (t < s) sbuf[t] += sbuf[t + s];
    __syncthreads();
  }
  float r = sbuf[0]; __syncthreads();
  return r;
}

__device__ inline float block_reduce_max(float v, float* sbuf) {
  int t = threadIdx.x;
  sbuf[t] = v; __syncthreads();
  for (int s = blockDim.x >> 1; s > 0; s >>= 1) {
    if (t < s) sbuf[t] = fmaxf(sbuf[t], sbuf[t + s]);
    __syncthreads();
  }
  float r = sbuf[0]; __syncthreads();
  return r;
}

// ------------------------------ kernels ------------------------------------

__global__ void k_gather_emb(const int* x, const float* embW, float* zf, float* vq_acc) {
  int m = blockIdx.x;
  const float* row = embW + (size_t)x[m] * TWOD;
  for (int k = threadIdx.x; k < TWOD; k += blockDim.x) zf[(size_t)m * TWOD + k] = row[k];
  if (m == 0 && threadIdx.x == 0) *vq_acc = 0.f;
}

__global__ void k_copy(float* dst, const float* src, int n) {
  int i = blockIdx.x * blockDim.x + threadIdx.x;
  if (i < n) dst[i] = src[i];
}

__global__ void k_cvt_bf16(const float* src, bf16_t* dst, int n) {
  int i = blockIdx.x * blockDim.x + threadIdx.x;
  if (i < n) dst[i] = (bf16_t)src[i];
}

__global__ void k_cvt_bf16_neg(const float* src, bf16_t* dst, int n) {
  int i = blockIdx.x * blockDim.x + threadIdx.x;
  if (i < n) dst[i] = (bf16_t)(-src[i]);
}

// dec_W [K=1024, N=32000] -> decT [N, K] bf16, tiled transpose through LDS.
__global__ void k_transcvt(const float* src, bf16_t* dst) {
  __shared__ float tile[32][33];
  int nb = blockIdx.x * 32, kb = blockIdx.y * 32;
  for (int i = threadIdx.y; i < 32; i += 8)
    tile[i][threadIdx.x] = src[(size_t)(kb + i) * V_ + nb + threadIdx.x];
  __syncthreads();
  for (int i = threadIdx.y; i < 32; i += 8)
    dst[(size_t)(nb + i) * TWOD + kb + threadIdx.x] = (bf16_t)tile[threadIdx.x][i];
}

__global__ void k_cb2(const float* cb, float* cb2) {
  __shared__ float sb[256];
  int v = blockIdx.x;
  float s = 0.f;
  for (int k = threadIdx.x; k < TWOD; k += 256) {
    float t = cb[(size_t)v * TWOD + k]; s += t * t;
  }
  s = block_reduce_sum(s, sb);
  if (threadIdx.x == 0) cb2[v] = s;
}

// zf -> bf16 copy + row sumsq + argmin reset. Run once per iteration (zf is
// only mutated by k_combine at iteration end) and once before the decoder.
__global__ void k_prepz(const float* zf, bf16_t* zfh, float* zf2, unsigned long long* amin) {
  __shared__ float sb[256];
  int m = blockIdx.x, t = threadIdx.x;
  const float* row = zf + (size_t)m * TWOD;
  float s = 0.f;
  for (int k = t; k < TWOD; k += 256) {
    float x = row[k]; s += x * x;
    zfh[(size_t)m * TWOD + k] = (bf16_t)x;
  }
  s = block_reduce_sum(s, sb);
  if (t == 0) { zf2[m] = s; amin[m] = ~0ull; }
}

// Six complex-linear outputs, all-bf16 operands; imaginary-weight sign folded
// into pre-negated bf16 weight copies. qfh = [qr|qi], kfh = [kr|ki], vfh = [vr|vi].
__global__ void k_clin(const bf16_t* zfh,
                       const bf16_t* wqr, const bf16_t* wqi, const bf16_t* wqin,
                       const bf16_t* wkr, const bf16_t* wki, const bf16_t* wkin,
                       const bf16_t* wvr, const bf16_t* wvi, const bf16_t* wvin,
                       bf16_t* qfh, bf16_t* kfh, bf16_t* vfh) {
  int nt = blockIdx.x, mt = blockIdx.y, job = blockIdx.z;
  const bf16_t* gr = zfh;
  const bf16_t* gi = zfh + D_;
  const bf16_t *A1, *A2, *W1, *W2; bf16_t* out; int ocol;
  switch (job) {
    case 0:  A1 = gr; W1 = wqr; A2 = gi; W2 = wqin; out = qfh; ocol = 0;   break;
    case 1:  A1 = gi; W1 = wqr; A2 = gr; W2 = wqi;  out = qfh; ocol = D_;  break;
    case 2:  A1 = gr; W1 = wkr; A2 = gi; W2 = wkin; out = kfh; ocol = 0;   break;
    case 3:  A1 = gi; W1 = wkr; A2 = gr; W2 = wki;  out = kfh; ocol = D_;  break;
    case 4:  A1 = gr; W1 = wvr; A2 = gi; W2 = wvin; out = vfh; ocol = 0;   break;
    default: A1 = gi; W1 = wvr; A2 = gr; W2 = wvi;  out = vfh; ocol = D_;  break;
  }
  v8f c = {};
  int row0 = mt * 16, n0 = nt * 16;
  for (int k0 = 0; k0 < D_; k0 += 32) {
    c = wmma_bf16(frag_rm_bf16(A1, TWOD, row0, k0), frag_rm_bf16(W1, D_, n0, k0), c);
  }
  for (int k0 = 0; k0 < D_; k0 += 32) {
    c = wmma_bf16(frag_rm_bf16(A2, TWOD, row0, k0), frag_rm_bf16(W2, D_, n0, k0), c);
  }
  store_tile_bf16(out, TWOD, row0, ocol + n0, c);
}

__global__ void k_scores(const bf16_t* qfh, const bf16_t* kfh, float* attn) {
  int nt = blockIdx.x, mt = blockIdx.y;
  v8f c = {};
  for (int k0 = 0; k0 < TWOD; k0 += 32) {
    c = wmma_bf16(frag_rm_bf16(qfh, TWOD, mt * 16, k0),
                  frag_rm_bf16(kfh, TWOD, nt * 16, k0), c);
  }
  store_tile(attn, B_, mt * 16, nt * 16, c, SCALE_, nullptr);
}

__global__ void k_softmax_rows(float* attn, int ncols) {
  __shared__ float sb[256];
  int m = blockIdx.x;
  float* row = attn + (size_t)m * ncols;
  float mx = -1e30f;
  for (int j = threadIdx.x; j < ncols; j += blockDim.x) mx = fmaxf(mx, row[j]);
  mx = block_reduce_max(mx, sb);
  float sum = 0.f;
  for (int j = threadIdx.x; j < ncols; j += blockDim.x) {
    float e = __expf(row[j] - mx); row[j] = e; sum += e;
  }
  sum = block_reduce_sum(sum, sb);
  float inv = 1.f / sum;
  for (int j = threadIdx.x; j < ncols; j += blockDim.x) row[j] *= inv;
}

__global__ void k_attnv(const float* attn, const bf16_t* vfh, float* flat) {
  int nt = blockIdx.x, mt = blockIdx.y, half = blockIdx.z;
  v8f c = {};
  for (int k0 = 0; k0 < B_; k0 += 32) {
    c = wmma_bf16(frag_rm_f32(attn, B_, mt * 16, k0),
                  frag_cm_bf16(vfh + half * D_, TWOD, nt * 16, k0), c);
  }
  store_tile(flat, TWOD, mt * 16, half * D_ + nt * 16, c, 1.f, nullptr);
}

__global__ void k_layernorm(float* flat, const float* g, const float* b) {
  __shared__ float sb[256];
  int m = blockIdx.x;
  float* row = flat + (size_t)m * TWOD;
  float s = 0.f;
  for (int k = threadIdx.x; k < TWOD; k += 256) s += row[k];
  float mean = block_reduce_sum(s, sb) * (1.f / TWOD);
  float v = 0.f;
  for (int k = threadIdx.x; k < TWOD; k += 256) { float d = row[k] - mean; v += d * d; }
  v = block_reduce_sum(v, sb) * (1.f / TWOD);
  float inv = rsqrtf(v + 1e-5f);
  for (int k = threadIdx.x; k < TWOD; k += 256)
    row[k] = (row[k] - mean) * inv * g[k] + b[k];
}

// out[m] = bias + sum_k x1[m,k]*W[k] (+ sum_k x2[m,k]*W[n1+k])
__global__ void k_saldot(const float* x1, int n1, const float* x2, int n2,
                         const float* W, const float* bias, float* out, int ld1, int ld2) {
  __shared__ float sb[256];
  int m = blockIdx.x;
  float s = 0.f;
  for (int k = threadIdx.x; k < n1; k += 256) s += x1[(size_t)m * ld1 + k] * W[k];
  if (x2)
    for (int k = threadIdx.x; k < n2; k += 256) s += x2[(size_t)m * ld2 + k] * W[n1 + k];
  s = block_reduce_sum(s, sb);
  if (threadIdx.x == 0) out[m] = s + bias[0];
}

__global__ void k_palette(const float* zf, const float* visP, const float* audP,
                          float* vresp, float* aresp) {
  __shared__ float wl[APAL];
  int m = blockIdx.x, which = blockIdx.y, t = threadIdx.x;
  const float* g = zf + (size_t)m * TWOD + which * D_;
  const float* P = which ? audP : visP;
  float* outp = (which ? aresp : vresp) + (size_t)m * D_;
  if (t < APAL) {
    float s = 0.f;
    for (int k = 0; k < D_; ++k) s += g[k] * P[(size_t)t * D_ + k];
    wl[t] = s;
  }
  __syncthreads();
  if (t == 0) {
    float mx = wl[0];
    for (int a = 1; a < APAL; ++a) mx = fmaxf(mx, wl[a]);
    float sum = 0.f;
    for (int a = 0; a < APAL; ++a) { wl[a] = __expf(wl[a] - mx); sum += wl[a]; }
    float inv = 1.f / sum;
    for (int a = 0; a < APAL; ++a) wl[a] *= inv;
  }
  __syncthreads();
  for (int d = t; d < D_; d += 256) {
    float s = 0.f;
    for (int a = 0; a < APAL; ++a) s += wl[a] * P[(size_t)a * D_ + d];
    outp[d] = s;
  }
}

__global__ void k_keyinv(const float* keys, float* keyinv) {
  __shared__ float sb[256];
  int s_ = blockIdx.x;
  float acc = 0.f;
  for (int k = threadIdx.x; k < D_; k += 256) {
    float t = keys[(size_t)s_ * D_ + k]; acc += t * t;
  }
  acc = block_reduce_sum(acc, sb);
  if (threadIdx.x == 0) keyinv[s_] = 1.f / fmaxf(sqrtf(acc), 1e-12f);
}

__global__ void k_memattn(const float* zf, const float* keys, const float* vals,
                          const float* keyinv, float* recalled, int S) {
  __shared__ float sb[256];
  __shared__ float slog[SMAX_];
  int m = blockIdx.x, t = threadIdx.x;
  const float* g = zf + (size_t)m * TWOD;  // gr half
  float acc = 0.f;
  for (int k = t; k < D_; k += 256) { float x = g[k]; acc += x * x; }
  acc = block_reduce_sum(acc, sb);
  float ginv = 1.f / fmaxf(sqrtf(acc), 1e-12f);
  for (int s_ = t; s_ < S; s_ += 256) {
    float dot = 0.f;
    const float* kr = keys + (size_t)s_ * D_;
    for (int k = 0; k < D_; ++k) dot += g[k] * kr[k];
    slog[s_] = 5.f * dot * ginv * keyinv[s_];
  }
  __syncthreads();
  float mx = -1e30f;
  for (int s_ = t; s_ < S; s_ += 256) mx = fmaxf(mx, slog[s_]);
  mx = block_reduce_max(mx, sb);
  float sum = 0.f;
  for (int s_ = t; s_ < S; s_ += 256) {
    float e = __expf(slog[s_] - mx); slog[s_] = e; sum += e;
  }
  sum = block_reduce_sum(sum, sb);
  float inv = 1.f / sum;
  for (int d = t; d < D_; d += 256) {
    float r = 0.f;
    for (int s_ = 0; s_ < S; ++s_) r += slog[s_] * vals[(size_t)s_ * D_ + d];
    recalled[(size_t)m * D_ + d] = r * inv;
  }
}

// keys/vals <- concat([gr, old[:-1]]) (ping-pong)
__global__ void k_memupdate(const float* zf, const float* okeys, const float* ovals,
                            float* nkeys, float* nvals) {
  int j = blockIdx.x, which = blockIdx.y;
  float* dst = (which ? nvals : nkeys) + (size_t)j * D_;
  const float* src = (j < B_) ? (zf + (size_t)j * TWOD)
                              : ((which ? ovals : okeys) + (size_t)(j - B_) * D_);
  for (int k = threadIdx.x; k < D_; k += blockDim.x) dst[k] = src[k];
}

// Stage 16 rows of bf16 A (32 KB) into LDS with async b128 copies.
__device__ inline void stage_A(const bf16_t* zfh, int mt, bf16_t* sA) {
  const char* gsrc = (const char*)(zfh + (size_t)mt * 16 * TWOD);
  uint32_t lbase = (uint32_t)(uintptr_t)(&sA[0]);
  for (int off = threadIdx.x * 16; off < 16 * TWOD * 2; off += blockDim.x * 16)
    async_b128(gsrc + off, lbase + (uint32_t)off);
  wait_async0();
  __syncthreads();
}

// --- VQ distance epilogue: pack (dist_bits << 32 | idx), atomicMin per row.
__device__ inline void vq_epilogue(v8f c, int mt, int vt, const float* zf2, const float* cb2,
                                   unsigned long long* amin) {
  int lane = threadIdx.x & 31;
  int r0 = mt * 16 + ((lane >> 4) << 3);
  int vv = vt + (lane & 15);
  float c2 = cb2[vv];
#pragma unroll
  for (int r = 0; r < 8; ++r) {
    int m = r0 + r;
    float d = fmaxf(zf2[m] + c2 - 2.f * c[r], 0.f);
    unsigned long long key =
        ((unsigned long long)__float_as_uint(d) << 32) | (unsigned int)vv;
    atomicMin(&amin[m], key);
  }
}

// Fast path: bf16 codebook, 4 waves/block, 4 N-tiles per wave (A reuse x4).
__global__ void k_vq4(const bf16_t* zfh, const bf16_t* cbh, const float* zf2, const float* cb2,
                      unsigned long long* amin) {
  __shared__ bf16_t sA[16 * TWOD];  // 32 KB
  int mt = blockIdx.y;
  stage_A(zfh, mt, sA);
  int wave = threadIdx.x >> 5;
  int vt = (blockIdx.x * 16 + wave * 4) * 16;
  v8f c0 = {}, c1 = {}, c2 = {}, c3 = {};
  for (int k0 = 0; k0 < TWOD; k0 += 32) {
    v16bf a  = frag_rm_bf16(sA, TWOD, 0, k0);
    v16bf b0 = frag_rm_bf16(cbh, TWOD, vt, k0);
    v16bf b1 = frag_rm_bf16(cbh, TWOD, vt + 16, k0);
    v16bf b2 = frag_rm_bf16(cbh, TWOD, vt + 32, k0);
    v16bf b3 = frag_rm_bf16(cbh, TWOD, vt + 48, k0);
    c0 = wmma_bf16(a, b0, c0);
    c1 = wmma_bf16(a, b1, c1);
    c2 = wmma_bf16(a, b2, c2);
    c3 = wmma_bf16(a, b3, c3);
  }
  vq_epilogue(c0, mt, vt,      zf2, cb2, amin);
  vq_epilogue(c1, mt, vt + 16, zf2, cb2, amin);
  vq_epilogue(c2, mt, vt + 32, zf2, cb2, amin);
  vq_epilogue(c3, mt, vt + 48, zf2, cb2, amin);
}

// Fallback: f32 codebook converted in-loop (used only if ws too small for cbh).
__global__ void k_vq(const bf16_t* zfh, const float* cb, const float* zf2, const float* cb2,
                     unsigned long long* amin) {
  __shared__ bf16_t sA[16 * TWOD];
  int mt = blockIdx.y;
  stage_A(zfh, mt, sA);
  int wave = threadIdx.x >> 5;
  int vt = (blockIdx.x * 8 + wave) * 16;
  v8f c = {};
  for (int k0 = 0; k0 < TWOD; k0 += 32) {
    c = wmma_bf16(frag_rm_bf16(sA, TWOD, 0, k0), frag_rm_f32(cb, TWOD, vt, k0), c);
  }
  vq_epilogue(c, mt, vt, zf2, cb2, amin);
}

__global__ void k_vqfinish(const unsigned long long* amin, const float* cb, const float* zf,
                           float* zq, float* vq_acc) {
  __shared__ float sb[256];
  int m = blockIdx.x, t = threadIdx.x;
  unsigned int idx = (unsigned int)(amin[m] & 0xffffffffu);
  const float* crow = cb + (size_t)idx * TWOD;
  const float* zrow = zf + (size_t)m * TWOD;
  float s = 0.f;
  for (int k = t; k < TWOD; k += 256) {
    float q = crow[k];
    zq[(size_t)m * TWOD + k] = q;
    float d = q - zrow[k];
    s += d * d;
  }
  s = block_reduce_sum(s, sb);
  if (t == 0) atomicAdd(vq_acc, 1.25f * s / (float)(B_ * TWOD));  // 1.0 + 0.25 commitment
}

__global__ void k_combine(float* zf, const float* flat, const float* vresp, const float* aresp,
                          const float* recalled, const float* zq,
                          const float* sp, const float* ss, const float* sm, const float* ssem) {
  int m = blockIdx.x, t = threadIdx.x;
  float l0 = sp[m] / TEMP_, l1 = ss[m] / TEMP_, l2 = sm[m] / TEMP_, l3 = ssem[m] / TEMP_;
  float mx = fmaxf(fmaxf(l0, l1), fmaxf(l2, l3));
  float e0 = __expf(l0 - mx), e1 = __expf(l1 - mx), e2 = __expf(l2 - mx), e3 = __expf(l3 - mx);
  float inv = 1.f / (e0 + e1 + e2 + e3);
  float w0 = e0 * inv, w1 = e1 * inv, w2 = e2 * inv, w3 = e3 * inv;
  size_t o = (size_t)m * TWOD, o5 = (size_t)m * D_;
  for (int d = t; d < D_; d += blockDim.x) {
    float nr = 0.6f * zf[o + d] +
               0.4f * (w0 * flat[o + d] + w1 * vresp[o5 + d] + w2 * recalled[o5 + d] + w3 * zq[o + d]);
    float ni = 0.6f * zf[o + D_ + d] +
               0.4f * (w0 * flat[o + D_ + d] + w1 * aresp[o5 + d] + w3 * zq[o + D_ + d]);
    zf[o + d] = nr;
    zf[o + D_ + d] = ni;
  }
}

// Fast decoder: bf16 transposed weights, 4 waves/block, 4 N-tiles per wave.
__global__ void k_dec4(const bf16_t* zfh, const bf16_t* decT, const float* decb, float* out) {
  __shared__ bf16_t sA[16 * TWOD];
  int mt = blockIdx.y;
  stage_A(zfh, mt, sA);
  int wave = threadIdx.x >> 5;
  int vt = (blockIdx.x * 16 + wave * 4) * 16;
  v8f c0 = {}, c1 = {}, c2 = {}, c3 = {};
  for (int k0 = 0; k0 < TWOD; k0 += 32) {
    v16bf a  = frag_rm_bf16(sA, TWOD, 0, k0);
    v16bf b0 = frag_rm_bf16(decT, TWOD, vt, k0);
    v16bf b1 = frag_rm_bf16(decT, TWOD, vt + 16, k0);
    v16bf b2 = frag_rm_bf16(decT, TWOD, vt + 32, k0);
    v16bf b3 = frag_rm_bf16(decT, TWOD, vt + 48, k0);
    c0 = wmma_bf16(a, b0, c0);
    c1 = wmma_bf16(a, b1, c1);
    c2 = wmma_bf16(a, b2, c2);
    c3 = wmma_bf16(a, b3, c3);
  }
  store_tile(out, V_, mt * 16, vt,      c0, 1.f, decb);
  store_tile(out, V_, mt * 16, vt + 16, c1, 1.f, decb);
  store_tile(out, V_, mt * 16, vt + 32, c2, 1.f, decb);
  store_tile(out, V_, mt * 16, vt + 48, c3, 1.f, decb);
}

// Fallback decoder: strided f32 column loads.
__global__ void k_decoder(const bf16_t* zfh, const float* decW, const float* decb, float* out) {
  int nt = blockIdx.x, mt = blockIdx.y;
  v8f c = {};
  for (int k0 = 0; k0 < TWOD; k0 += 32) {
    c = wmma_bf16(frag_rm_bf16(zfh, TWOD, mt * 16, k0),
                  frag_cm_f32(decW, V_, nt * 16, k0), c);
  }
  store_tile(out, V_, mt * 16, nt * 16, c, 1.f, decb);
}

__global__ void k_epilogue(const float* vq_acc, float* out, int out_size) {
  if (blockIdx.x == 0 && threadIdx.x == 0) {
    if (out_size > B_ * V_)     out[B_ * V_] = *vq_acc;
    if (out_size > B_ * V_ + 1) out[B_ * V_ + 1] = (float)MAXREC_;
  }
}

// ------------------------------ host launch --------------------------------

extern "C" void kernel_launch(void* const* d_in, const int* in_sizes, int n_in,
                              void* d_out, int out_size, void* d_ws, size_t ws_size,
                              hipStream_t stream) {
  (void)in_sizes; (void)n_in;
  const int*   x        = (const int*)  d_in[0];
  const float* embW     = (const float*)d_in[1];
  const float* Wqr      = (const float*)d_in[2];
  const float* Wqi      = (const float*)d_in[3];
  const float* Wkr      = (const float*)d_in[4];
  const float* Wki      = (const float*)d_in[5];
  const float* Wvr      = (const float*)d_in[6];
  const float* Wvi      = (const float*)d_in[7];
  const float* ln_g     = (const float*)d_in[8];
  const float* ln_b     = (const float*)d_in[9];
  const float* procW    = (const float*)d_in[10];
  const float* procB    = (const float*)d_in[11];
  const float* visP     = (const float*)d_in[12];
  const float* audP     = (const float*)d_in[13];
  const float* sensW    = (const float*)d_in[14];
  const float* sensB    = (const float*)d_in[15];
  const float* memKeys  = (const float*)d_in[16];
  const float* memVals  = (const float*)d_in[17];
  const float* memW     = (const float*)d_in[18];
  const float* memB     = (const float*)d_in[19];
  const float* codebook = (const float*)d_in[20];
  const float* semW     = (const float*)d_in[21];
  const float* semB     = (const float*)d_in[22];
  const float* decW     = (const float*)d_in[23];
  const float* decb     = (const float*)d_in[24];
  float* out = (float*)d_out;

  char* w = (char*)d_ws;
  auto alloc = [&](size_t bytes) -> char* {
    char* p = w; w += (bytes + 255) & ~(size_t)255; return p;
  };
  float*  zf       = (float*)  alloc((size_t)B_ * TWOD * 4);
  bf16_t* zfh      = (bf16_t*) alloc((size_t)B_ * TWOD * 2);
  bf16_t* qfh      = (bf16_t*) alloc((size_t)B_ * TWOD * 2);
  bf16_t* kfh      = (bf16_t*) alloc((size_t)B_ * TWOD * 2);
  bf16_t* vfh      = (bf16_t*) alloc((size_t)B_ * TWOD * 2);
  float*  attn     = (float*)  alloc((size_t)B_ * B_ * 4);
  float*  flat     = (float*)  alloc((size_t)B_ * TWOD * 4);
  float*  vresp    = (float*)  alloc((size_t)B_ * D_ * 4);
  float*  aresp    = (float*)  alloc((size_t)B_ * D_ * 4);
  float*  recalled = (float*)  alloc((size_t)B_ * D_ * 4);
  float*  sp       = (float*)  alloc(B_ * 4);
  float*  ss       = (float*)  alloc(B_ * 4);
  float*  sm       = (float*)  alloc(B_ * 4);
  float*  ssem     = (float*)  alloc(B_ * 4);
  float*  keyinv   = (float*)  alloc(SMAX_ * 4);
  float*  keysA    = (float*)  alloc((size_t)SMAX_ * D_ * 4);
  float*  keysB    = (float*)  alloc((size_t)SMAX_ * D_ * 4);
  float*  valsA    = (float*)  alloc((size_t)SMAX_ * D_ * 4);
  float*  valsB    = (float*)  alloc((size_t)SMAX_ * D_ * 4);
  float*  zq       = (float*)  alloc((size_t)B_ * TWOD * 4);
  float*  zf2      = (float*)  alloc(B_ * 4);
  unsigned long long* amin = (unsigned long long*)alloc(B_ * 8);
  float*  vq_acc   = (float*)  alloc(256);
  float*  cb2      = (float*)  alloc((size_t)V_ * 4);
  // bf16 projection weights (+ negated imaginary copies), 0.5 MB each.
  const int WN = D_ * D_;
  bf16_t* wqrh  = (bf16_t*)alloc((size_t)WN * 2);
  bf16_t* wqih  = (bf16_t*)alloc((size_t)WN * 2);
  bf16_t* wqinh = (bf16_t*)alloc((size_t)WN * 2);
  bf16_t* wkrh  = (bf16_t*)alloc((size_t)WN * 2);
  bf16_t* wkih  = (bf16_t*)alloc((size_t)WN * 2);
  bf16_t* wkinh = (bf16_t*)alloc((size_t)WN * 2);
  bf16_t* wvrh  = (bf16_t*)alloc((size_t)WN * 2);
  bf16_t* wvih  = (bf16_t*)alloc((size_t)WN * 2);
  bf16_t* wvinh = (bf16_t*)alloc((size_t)WN * 2);

  // Optional bf16 weight caches (fast path) -- only if ws is large enough.
  bf16_t* cbh  = (bf16_t*)alloc((size_t)V_ * TWOD * 2);
  bool have_cbh = ((size_t)(w - (char*)d_ws) <= ws_size);
  bf16_t* decT = (bf16_t*)alloc((size_t)V_ * TWOD * 2);
  bool have_dect = ((size_t)(w - (char*)d_ws) <= ws_size);

  k_gather_emb<<<B_, 256, 0, stream>>>(x, embW, zf, vq_acc);
  k_copy<<<(S0_ * D_ + 255) / 256, 256, 0, stream>>>(keysA, memKeys, S0_ * D_);
  k_copy<<<(S0_ * D_ + 255) / 256, 256, 0, stream>>>(valsA, memVals, S0_ * D_);
  k_cb2<<<V_, 256, 0, stream>>>(codebook, cb2);
  const int wgrid = (WN + 255) / 256;
  k_cvt_bf16    <<<wgrid, 256, 0, stream>>>(Wqr, wqrh, WN);
  k_cvt_bf16    <<<wgrid, 256, 0, stream>>>(Wqi, wqih, WN);
  k_cvt_bf16_neg<<<wgrid, 256, 0, stream>>>(Wqi, wqinh, WN);
  k_cvt_bf16    <<<wgrid, 256, 0, stream>>>(Wkr, wkrh, WN);
  k_cvt_bf16    <<<wgrid, 256, 0, stream>>>(Wki, wkih, WN);
  k_cvt_bf16_neg<<<wgrid, 256, 0, stream>>>(Wki, wkinh, WN);
  k_cvt_bf16    <<<wgrid, 256, 0, stream>>>(Wvr, wvrh, WN);
  k_cvt_bf16    <<<wgrid, 256, 0, stream>>>(Wvi, wvih, WN);
  k_cvt_bf16_neg<<<wgrid, 256, 0, stream>>>(Wvi, wvinh, WN);
  if (have_cbh)
    k_cvt_bf16<<<(V_ * TWOD + 255) / 256, 256, 0, stream>>>(codebook, cbh, V_ * TWOD);
  if (have_dect)
    k_transcvt<<<dim3(V_ / 32, TWOD / 32), dim3(32, 8), 0, stream>>>(decW, decT);

  float* kcur = keysA; float* knxt = keysB;
  float* vcur = valsA; float* vnxt = valsB;
  int S = S0_;
  for (int r = 0; r < MAXREC_; ++r) {
    k_prepz<<<B_, 256, 0, stream>>>(zf, zfh, zf2, amin);
    k_clin<<<dim3(D_ / 16, B_ / 16, 6), 32, 0, stream>>>(
        zfh, wqrh, wqih, wqinh, wkrh, wkih, wkinh, wvrh, wvih, wvinh, qfh, kfh, vfh);
    k_scores<<<dim3(B_ / 16, B_ / 16), 32, 0, stream>>>(qfh, kfh, attn);
    k_softmax_rows<<<B_, 128, 0, stream>>>(attn, B_);
    k_attnv<<<dim3(D_ / 16, B_ / 16, 2), 32, 0, stream>>>(attn, vfh, flat);
    k_layernorm<<<B_, 256, 0, stream>>>(flat, ln_g, ln_b);
    k_saldot<<<B_, 256, 0, stream>>>(flat, TWOD, (const float*)nullptr, 0,
                                     procW, procB, sp, TWOD, 0);
    k_palette<<<dim3(B_, 2), 256, 0, stream>>>(zf, visP, audP, vresp, aresp);
    k_saldot<<<B_, 256, 0, stream>>>(vresp, D_, aresp, D_, sensW, sensB, ss, D_, D_);
    k_keyinv<<<S, 256, 0, stream>>>(kcur, keyinv);
    k_memattn<<<B_, 256, 0, stream>>>(zf, kcur, vcur, keyinv, recalled, S);
    k_saldot<<<B_, 256, 0, stream>>>(recalled, D_, (const float*)nullptr, 0,
                                     memW, memB, sm, D_, 0);
    int Snew = S + B_ - 1; if (Snew > SMAX_) Snew = SMAX_;
    k_memupdate<<<dim3(Snew, 2), 128, 0, stream>>>(zf, kcur, vcur, knxt, vnxt);
    if (have_cbh)
      k_vq4<<<dim3(V_ / 256, B_ / 16), 128, 0, stream>>>(zfh, cbh, zf2, cb2, amin);
    else
      k_vq<<<dim3(V_ / 128, B_ / 16), 256, 0, stream>>>(zfh, codebook, zf2, cb2, amin);
    k_vqfinish<<<B_, 256, 0, stream>>>(amin, codebook, zf, zq, vq_acc);
    k_saldot<<<B_, 256, 0, stream>>>(zq, TWOD, (const float*)nullptr, 0,
                                     semW, semB, ssem, TWOD, 0);
    k_combine<<<B_, 256, 0, stream>>>(zf, flat, vresp, aresp, recalled, zq, sp, ss, sm, ssem);
    { float* t = kcur; kcur = knxt; knxt = t; t = vcur; vcur = vnxt; vnxt = t; }
    S = Snew;
  }
  k_prepz<<<B_, 256, 0, stream>>>(zf, zfh, zf2, amin);  // refresh bf16 zf for decoder
  if (have_dect)
    k_dec4<<<dim3(V_ / 256, B_ / 16), 128, 0, stream>>>(zfh, decT, decb, out);
  else
    k_decoder<<<dim3(V_ / 16, B_ / 16), 32, 0, stream>>>(zfh, decW, decb, out);
  k_epilogue<<<1, 32, 0, stream>>>(vq_acc, out, out_size);
}
